// Attention_66005057404975
// MI455X (gfx1250) — compile-verified, matrix operand order checked by reading, not run on hardware
//
#include <hip/hip_runtime.h>
#include <stdint.h>

// ---------------------------------------------------------------------------
// Problem constants (from reference): B=64 N=289 C=768 H=12 D=64 P=1 L=32
// ---------------------------------------------------------------------------
#define BB   64
#define NN   289
#define CC   768
#define HH   12
#define DD   64
#define NIMG 256           // N - P - L
#define NPADV 320          // padded column count for V^T rows (16B aligned)
#define MROWS (BB*NN)      // 18496 = 578 * 32 exactly
#define QTILES 19          // ceil(289/16)
#define JCHUNKS 10         // ceil(289/32)

typedef __attribute__((ext_vector_type(16))) __bf16 v16bf;
typedef __attribute__((ext_vector_type(8)))  __bf16 v8bf;
typedef __attribute__((ext_vector_type(8)))  float  v8f;

// ---------------- bf16 <-> f32 helpers (bit-level, RNE) ---------------------
__device__ __forceinline__ uint16_t f2bf_bits(float f) {
  union { float f; uint32_t u; } v; v.f = f;
  uint32_t r = v.u + 0x7FFFu + ((v.u >> 16) & 1u);
  return (uint16_t)(r >> 16);
}
__device__ __forceinline__ float bf2f(uint16_t b) {
  union { uint32_t u; float f; } v; v.u = ((uint32_t)b) << 16; return v.f;
}

// ---------------- WMMA wrapper ---------------------------------------------
__device__ __forceinline__ v8f wmma_bf16(v16bf a, v16bf b, v8f c) {
  return __builtin_amdgcn_wmma_f32_16x16x32_bf16(
      /*neg_a=*/false, a, /*neg_b=*/false, b,
      /*c_mod=*/(short)0, c, /*reuse_a=*/false, /*reuse_b=*/false);
}

// A-style fragment load (16x32 M x K, 16-bit elems) from row-major bf16.
// lane: row = (lane&15); hi = lane>>4 selects K subsets {8h..8h+7} and
// {16+8h..16+8h+7} per ISA 7.12.2.  Also used for B operands whenever the
// K x N matrix is stored N-major (we load rows of B^T).
__device__ __forceinline__ v16bf load_a_bf16(const uint16_t* base, int ld,
                                             int row, int kb, int hi) {
  const uint16_t* p = base + (size_t)row * ld + kb + 8 * hi;
  v8bf x0 = *reinterpret_cast<const v8bf*>(p);
  v8bf x1 = *reinterpret_cast<const v8bf*>(p + 16);
  v16bf r;
#pragma unroll
  for (int i = 0; i < 8; ++i) { r[i] = x0[i]; r[8 + i] = x1[i]; }
  return r;
}

// ---------------------------------------------------------------------------
// Kernel 0a: one-shot fp32 -> bf16 conversion (8 elements / thread).
// Keeps RNE-convert VALU out of the GEMM inner loops.
// ---------------------------------------------------------------------------
__global__ __launch_bounds__(256) void cvt_bf16_kernel(
    const float* __restrict__ src, uint16_t* __restrict__ dst, int n8) {
  int i = blockIdx.x * 256 + threadIdx.x;
  if (i >= n8) return;
  const float4* p = reinterpret_cast<const float4*>(src) + 2 * (size_t)i;
  float4 a = p[0], b = p[1];
  uint4 o;
  o.x = (uint32_t)f2bf_bits(a.x) | ((uint32_t)f2bf_bits(a.y) << 16);
  o.y = (uint32_t)f2bf_bits(a.z) | ((uint32_t)f2bf_bits(a.w) << 16);
  o.z = (uint32_t)f2bf_bits(b.x) | ((uint32_t)f2bf_bits(b.y) << 16);
  o.w = (uint32_t)f2bf_bits(b.z) | ((uint32_t)f2bf_bits(b.w) << 16);
  reinterpret_cast<uint4*>(dst)[(size_t)i] = o;
}

// Kernel 0b: zero the 31 pad columns of every V^T row (cols 289..319).
__global__ __launch_bounds__(64) void vpad_kernel(uint16_t* __restrict__ vT) {
  const size_t row = (size_t)blockIdx.x * DD + threadIdx.x;  // bh*64 + d
  uint16_t* p = vT + row * NPADV + NN;
#pragma unroll
  for (int i = 0; i < NPADV - NN; ++i) p[i] = 0;
}

// ---------------------------------------------------------------------------
// Kernel 1: FUSED qkv GEMM + LayerNorm + RoPE + q-scale.
// qkv = x @ qkv_w^T  (M=18496, K=768, N=2304).  A 64-wide output column
// block is exactly one (which,head): which = nt/12, h = nt%12, so each
// wave's 32x64 tile holds 32 tokens x full head-dim and LN can be done
// in-register: row stats via shfl-xor over the 16-lane C/D halves, RoPE
// pair exchange via shfl-xor(1) (pair = adjacent lanes).
// Outputs: q,k row-major (B,H,N,64) bf16;  V transposed (B,H,64,NPADV) bf16.
// ---------------------------------------------------------------------------
__global__ __launch_bounds__(128) void qkv_fused_kernel(
    const uint16_t* __restrict__ xB, const uint16_t* __restrict__ wB,
    const float* __restrict__ rope,
    const float* __restrict__ qg, const float* __restrict__ qb,
    const float* __restrict__ kg, const float* __restrict__ kbeta,
    uint16_t* __restrict__ qO, uint16_t* __restrict__ kO,
    uint16_t* __restrict__ vT) {
  const int lane = threadIdx.x & 31;
  const int wave = threadIdx.x >> 5;
  const int hi = lane >> 4, col = lane & 15;
  const int tile = blockIdx.x * 4 + wave;          // 578 * 36 tiles exactly
  const int mt = tile / 36, nt = tile % 36;
  const int mBase = mt * 32, nBase = nt * 64;
  const int which = nt / HH, h = nt % HH;          // 0=q 1=k 2=v

  v8f acc[2][4] = {{v8f{}, v8f{}, v8f{}, v8f{}}, {v8f{}, v8f{}, v8f{}, v8f{}}};
  for (int kb = 0; kb < CC; kb += 32) {
    if (kb + 32 < CC)
      __builtin_prefetch(xB + (size_t)(mBase + col) * CC + kb + 32, 0, 0);
    v16bf a0 = load_a_bf16(xB, CC, mBase + col, kb, hi);
    v16bf a1 = load_a_bf16(xB, CC, mBase + 16 + col, kb, hi);
#pragma unroll
    for (int t = 0; t < 4; ++t) {
      v16bf b = load_a_bf16(wB, CC, nBase + 16 * t + col, kb, hi);
      acc[0][t] = wmma_bf16(a0, b, acc[0][t]);
      acc[1][t] = wmma_bf16(a1, b, acc[1][t]);
    }
  }

  // Per-column LN gamma/beta (column c = 16t+col is head-dim index).
  float gv[4], bv[4];
  if (which < 2) {
    const float* g  = (which == 0) ? qg : kg;
    const float* be = (which == 0) ? qb : kbeta;
#pragma unroll
    for (int t = 0; t < 4; ++t) { gv[t] = g[16 * t + col]; bv[t] = be[16 * t + col]; }
  }

#pragma unroll
  for (int gr = 0; gr < 2; ++gr) {
#pragma unroll
    for (int jj = 0; jj < 8; ++jj) {
      const int m = mBase + 16 * gr + 8 * hi + jj;   // global token row
      const int bidx = m / NN, n = m % NN;
      float vals[4];
#pragma unroll
      for (int t = 0; t < 4; ++t) vals[t] = acc[gr][t][jj];

      if (which < 2) {
        // --- LayerNorm over D=64 (this row spans 4 cols x 16 lanes) ---
        float s  = vals[0] + vals[1] + vals[2] + vals[3];
        float ss = vals[0]*vals[0] + vals[1]*vals[1] +
                   vals[2]*vals[2] + vals[3]*vals[3];
        s  += __shfl_xor(s, 1, 32);  ss += __shfl_xor(ss, 1, 32);
        s  += __shfl_xor(s, 2, 32);  ss += __shfl_xor(ss, 2, 32);
        s  += __shfl_xor(s, 4, 32);  ss += __shfl_xor(ss, 4, 32);
        s  += __shfl_xor(s, 8, 32);  ss += __shfl_xor(ss, 8, 32);
        const float mean = s * (1.0f / 64.0f);
        const float inv  = rsqrtf(ss * (1.0f / 64.0f) - mean * mean + 1e-5f);
#pragma unroll
        for (int t = 0; t < 4; ++t)
          vals[t] = (vals[t] - mean) * inv * gv[t] + bv[t];

        // --- RoPE on image tokens n in [1,257); pair = adjacent lanes ---
        if (n >= 1 && n < 1 + NIMG) {
          const float* rp = rope + (size_t)(n - 1) * DD;
#pragma unroll
          for (int t = 0; t < 4; ++t) {
            const int c = 16 * t + col;
            const float cs = rp[c & ~1], sn = rp[(c & ~1) + 1];
            const float partner = __shfl_xor(vals[t], 1, 32);
            vals[t] = (col & 1) ? (vals[t] * cs + partner * sn)
                                : (vals[t] * cs - partner * sn);
          }
        }
        if (which == 0) {
#pragma unroll
          for (int t = 0; t < 4; ++t) vals[t] *= 0.125f;  // D^-0.5
        }
        uint16_t* dst = ((which == 0) ? qO : kO) +
                        ((size_t)(bidx * HH + h) * NN + n) * DD;
#pragma unroll
        for (int t = 0; t < 4; ++t) dst[16 * t + col] = f2bf_bits(vals[t]);
      } else {
        // --- V: store transposed (B,H,64,NPADV) for P@V B-fragments ---
        uint16_t* dst = vT + (size_t)(bidx * HH + h) * DD * NPADV;
#pragma unroll
        for (int t = 0; t < 4; ++t)
          dst[(size_t)(16 * t + col) * NPADV + n] = f2bf_bits(vals[t]);
      }
    }
  }
}

// ---------------------------------------------------------------------------
// Kernel 2: flash attention.  One wave per (b,h, 16-row q-tile).
// S = Q K^T via 2 chained WMMAs (K-dim 64); online softmax in C/D layout
// (lane holds 8 rows, cols across 16-lane halves -> shfl-xor reductions);
// P staged through LDS to become an A fragment; O += P V via WMMA from the
// transposed V buffer.
// ---------------------------------------------------------------------------
__global__ __launch_bounds__(128) void attn_kernel(
    const uint16_t* __restrict__ qB, const uint16_t* __restrict__ kB,
    const uint16_t* __restrict__ vTB, uint16_t* __restrict__ aOut) {
  __shared__ __align__(16) uint16_t ldsP[4][16 * 32];  // per-wave P tile
  const int lane = threadIdx.x & 31;
  const int wave = threadIdx.x >> 5;
  const int hi = lane >> 4, col = lane & 15;
  const int w = blockIdx.x * 4 + wave;                 // 768*19 waves exactly
  const int bh = w / QTILES, qt = w % QTILES;

  const uint16_t* qp = qB  + (size_t)bh * NN * DD;
  const uint16_t* kp = kB  + (size_t)bh * NN * DD;
  const uint16_t* vp = vTB + (size_t)bh * DD * NPADV;

  int qrow = qt * 16 + col; if (qrow > NN - 1) qrow = NN - 1;
  v16bf qa0 = load_a_bf16(qp, DD, qrow, 0,  hi);
  v16bf qa1 = load_a_bf16(qp, DD, qrow, 32, hi);

  float mrow[8], lrow[8];
  v8f oacc[4] = {v8f{}, v8f{}, v8f{}, v8f{}};
#pragma unroll
  for (int jj = 0; jj < 8; ++jj) { mrow[jj] = -1e30f; lrow[jj] = 0.f; }

  for (int cbk = 0; cbk < JCHUNKS; ++cbk) {
    const int jBase = cbk * 32;
    v8f sfr[2];
#pragma unroll
    for (int tt = 0; tt < 2; ++tt) {
      int krow = jBase + 16 * tt + col; if (krow > NN - 1) krow = NN - 1;
      v16bf kb0 = load_a_bf16(kp, DD, krow, 0,  hi);
      v16bf kb1 = load_a_bf16(kp, DD, krow, 32, hi);
      v8f s = v8f{};
      s = wmma_bf16(qa0, kb0, s);
      s = wmma_bf16(qa1, kb1, s);
      sfr[tt] = s;
    }
    const bool oob0 = (jBase + col)      > NN - 1;
    const bool oob1 = (jBase + 16 + col) > NN - 1;
#pragma unroll
    for (int jj = 0; jj < 8; ++jj) {
      float s0 = oob0 ? -1e30f : sfr[0][jj];
      float s1 = oob1 ? -1e30f : sfr[1][jj];
      float mx = fmaxf(s0, s1);
      mx = fmaxf(mx, __shfl_xor(mx, 1, 32));
      mx = fmaxf(mx, __shfl_xor(mx, 2, 32));
      mx = fmaxf(mx, __shfl_xor(mx, 4, 32));
      mx = fmaxf(mx, __shfl_xor(mx, 8, 32));
      float mnew  = fmaxf(mrow[jj], mx);
      float scale = __expf(mrow[jj] - mnew);
      float p0 = __expf(s0 - mnew);
      float p1 = __expf(s1 - mnew);
      float rs = p0 + p1;
      rs += __shfl_xor(rs, 1, 32);
      rs += __shfl_xor(rs, 2, 32);
      rs += __shfl_xor(rs, 4, 32);
      rs += __shfl_xor(rs, 8, 32);
      lrow[jj] = lrow[jj] * scale + rs;
      mrow[jj] = mnew;
#pragma unroll
      for (int t = 0; t < 4; ++t) oacc[t][jj] *= scale;
      const int prow = 8 * hi + jj;
      ldsP[wave][prow * 32 + col]      = f2bf_bits(p0);
      ldsP[wave][prow * 32 + 16 + col] = f2bf_bits(p1);
    }
    // Re-enter P as an A fragment (16x32) from LDS, accumulate O += P V.
    v16bf pa = load_a_bf16(&ldsP[wave][0], 32, col, 0, hi);
#pragma unroll
    for (int t = 0; t < 4; ++t) {
      v16bf vb = load_a_bf16(vp, NPADV, 16 * t + col, jBase, hi);
      oacc[t] = wmma_bf16(pa, vb, oacc[t]);
    }
  }

  // Epilogue: O /= l, write (B,N,C) bf16 for the projection GEMM.
  const int b = bh / HH, h = bh % HH;
#pragma unroll
  for (int jj = 0; jj < 8; ++jj) {
    const int ntok = qt * 16 + 8 * hi + jj;
    if (ntok >= NN) continue;
    const float invl = 1.0f / lrow[jj];
    uint16_t* orow = aOut + ((size_t)b * NN + ntok) * CC + h * DD;
#pragma unroll
    for (int t = 0; t < 4; ++t)
      orow[16 * t + col] = f2bf_bits(oacc[t][jj] * invl);
  }
}

// ---------------------------------------------------------------------------
// Kernel 3: out = attn_out @ proj_w^T + proj_b  (M=18496, K=768, N=768).
// 32x64 tile per wave: two A fragments share every B fragment.
// ---------------------------------------------------------------------------
__global__ __launch_bounds__(128) void proj_gemm_kernel(
    const uint16_t* __restrict__ aB, const uint16_t* __restrict__ wB,
    const float* __restrict__ bias, float* __restrict__ out) {
  const int lane = threadIdx.x & 31;
  const int wave = threadIdx.x >> 5;
  const int hi = lane >> 4, col = lane & 15;
  const int tile = blockIdx.x * 4 + wave;          // 578 * 12 tiles exactly
  const int mt = tile / 12, nt = tile % 12;
  const int mBase = mt * 32, nBase = nt * 64;

  v8f acc[2][4] = {{v8f{}, v8f{}, v8f{}, v8f{}}, {v8f{}, v8f{}, v8f{}, v8f{}}};
  for (int kb = 0; kb < CC; kb += 32) {
    v16bf a0 = load_a_bf16(aB, CC, mBase + col, kb, hi);
    v16bf a1 = load_a_bf16(aB, CC, mBase + 16 + col, kb, hi);
#pragma unroll
    for (int t = 0; t < 4; ++t) {
      v16bf b = load_a_bf16(wB, CC, nBase + 16 * t + col, kb, hi);
      acc[0][t] = wmma_bf16(a0, b, acc[0][t]);
      acc[1][t] = wmma_bf16(a1, b, acc[1][t]);
    }
  }
#pragma unroll
  for (int t = 0; t < 4; ++t) {
    const float bi = bias[nBase + 16 * t + col];
#pragma unroll
    for (int gr = 0; gr < 2; ++gr)
#pragma unroll
      for (int jj = 0; jj < 8; ++jj)
        out[(size_t)(mBase + 16 * gr + 8 * hi + jj) * CC + nBase + 16 * t + col] =
            acc[gr][t][jj] + bi;
  }
}

// ---------------------------------------------------------------------------
extern "C" void kernel_launch(void* const* d_in, const int* in_sizes, int n_in,
                              void* d_out, int out_size, void* d_ws,
                              size_t ws_size, hipStream_t stream) {
  (void)in_sizes; (void)n_in; (void)out_size; (void)ws_size;
  const float* x      = (const float*)d_in[0];
  const float* rope   = (const float*)d_in[1];
  const float* qkv_w  = (const float*)d_in[2];
  const float* proj_w = (const float*)d_in[3];
  const float* proj_b = (const float*)d_in[4];
  const float* qn_g   = (const float*)d_in[5];
  const float* qn_b   = (const float*)d_in[6];
  const float* kn_g   = (const float*)d_in[7];
  const float* kn_b   = (const float*)d_in[8];

  char* ws = (char*)d_ws;
  const size_t sz_x  = (size_t)MROWS * CC * 2;            // 28.4 MB
  const size_t sz_wq = (size_t)3 * CC * CC * 2;           //  3.5 MB
  const size_t sz_wp = (size_t)CC * CC * 2;               //  1.2 MB
  const size_t sz_qk = (size_t)BB * HH * NN * DD * 2;     // 28.4 MB
  const size_t sz_vT = (size_t)BB * HH * DD * NPADV * 2;  // 31.5 MB
  uint16_t* xB   = (uint16_t*)(ws);
  uint16_t* wqB  = (uint16_t*)(ws + sz_x);
  uint16_t* wpB  = (uint16_t*)(ws + sz_x + sz_wq);
  uint16_t* qBuf = (uint16_t*)(ws + sz_x + sz_wq + sz_wp);
  uint16_t* kBuf = (uint16_t*)(ws + sz_x + sz_wq + sz_wp + sz_qk);
  uint16_t* vT   = (uint16_t*)(ws + sz_x + sz_wq + sz_wp + 2 * sz_qk);
  uint16_t* attB = (uint16_t*)(ws + sz_x + sz_wq + sz_wp + 2 * sz_qk + sz_vT);

  // 0) one-shot bf16 conversions + V^T pad zeroing
  cvt_bf16_kernel<<<(MROWS * CC / 8 + 255) / 256, 256, 0, stream>>>(x, xB, MROWS * CC / 8);
  cvt_bf16_kernel<<<(3 * CC * CC / 8 + 255) / 256, 256, 0, stream>>>(qkv_w, wqB, 3 * CC * CC / 8);
  cvt_bf16_kernel<<<(CC * CC / 8 + 255) / 256, 256, 0, stream>>>(proj_w, wpB, CC * CC / 8);
  vpad_kernel<<<BB * HH, DD, 0, stream>>>(vT);

  // 1) fused QKV GEMM + LN + RoPE: 578*36 wave-tiles / 4 waves per block
  qkv_fused_kernel<<<5202, 128, 0, stream>>>(xB, wqB, rope, qn_g, qn_b,
                                             kn_g, kn_b, qBuf, kBuf, vT);
  // 2) flash attention: 768*19 wave-tiles / 4 waves per block
  attn_kernel<<<3648, 128, 0, stream>>>(qBuf, kBuf, vT, attB);
  // 3) output projection: 578*12 wave-tiles / 4 waves per block
  proj_gemm_kernel<<<1734, 128, 0, stream>>>(attB, wpB, proj_b,
                                             (float*)d_out);
}
